// BackboneMPNN_35699768165189
// MI455X (gfx1250) — matrix operand order repaired and checked by reading, not compiled, not run on hardware
//
#include <hip/hip_runtime.h>
#include <hip/hip_bf16.h>
#include <math.h>

// ---------------------------------------------------------------------------
// NeuroSAT-style MPNN on gfx1250 (MI455X), wave32 + WMMA bf16 16x16x32.
//   h state: [NV+NC][64] bf16 in workspace (halves the dominant gather traffic;
//            all consumers are bf16 WMMA A-operands anyway)
//   agg    : [max(NC,NV)][64] f32 scatter buffer (atomic f32 adds)
// Edge MLP / GRU / projection all run through v_wmma_f32_16x16x32_bf16 with
// weights staged once per block into LDS as bf16 (B[k][n]=w[n][k] contiguous).
// ---------------------------------------------------------------------------

#define HD    64
#define NV    100000
#define NC    300000
#define NEDGE 900000
#define NODES (NV + NC)

typedef __attribute__((ext_vector_type(16))) __bf16 v16bf;
typedef __attribute__((ext_vector_type(8)))  float  v8f;

__device__ __forceinline__ v8f wmma_bf16(v16bf a, v16bf b, v8f c) {
  return __builtin_amdgcn_wmma_f32_16x16x32_bf16(
      /*neg_a=*/false, a, /*neg_b=*/false, b,
      /*c_mod=*/(short)0, c, /*reuse_a=*/false, /*reuse_b=*/false);
}

// A fragment (16x32 bf16, M=row = lane%16) from a bf16 row of 64 elems.
// Layout: lane<16 -> K = s*32 + {0..7, 16..23}; lane>=16 -> +8. Two 16B loads.
__device__ __forceinline__ v16bf a_frag_bf16(const __bf16* __restrict__ row,
                                             int s, int lane) {
  const int k0 = s * 32 + ((lane & 16) ? 8 : 0);
  v16bf a;
#pragma unroll
  for (int i = 0; i < 8; ++i) a[i]     = row[k0 + i];
#pragma unroll
  for (int i = 0; i < 8; ++i) a[i + 8] = row[k0 + 16 + i];
  return a;
}

// Same fragment but sourced from an f32 row (used for the f32 agg buffer).
__device__ __forceinline__ v16bf a_frag_f32(const float* __restrict__ row,
                                            int s, int lane) {
  const int k0 = s * 32 + ((lane & 16) ? 8 : 0);
  v16bf a;
#pragma unroll
  for (int i = 0; i < 8; ++i) a[i]     = (__bf16)row[k0 + i];
#pragma unroll
  for (int i = 0; i < 8; ++i) a[i + 8] = (__bf16)row[k0 + 16 + i];
  return a;
}

// B fragment (32x16 bf16) from LDS weights stored row-major [n][k] (stride k).
// B[k][n] = w[n][k]; lane<16: K=kbase+{0..7,16..23}, lane>=16: +8.
__device__ __forceinline__ v16bf b_frag(const __bf16* __restrict__ w, int stride,
                                        int ntile, int kbase, int lane) {
  const int n    = ntile * 16 + (lane & 15);
  const int koff = kbase + ((lane & 16) ? 8 : 0);
  const __bf16* p = w + n * stride + koff;
  v16bf b;
#pragma unroll
  for (int i = 0; i < 8; ++i) { b[i] = p[i]; b[i + 8] = p[16 + i]; }
  return b;
}

__device__ __forceinline__ float sigmoidf_(float x) {
  return 1.0f / (1.0f + __expf(-x));
}

// ------------------------------ init / zero --------------------------------
__global__ __launch_bounds__(256) void init_h_kernel(__bf16* __restrict__ hb,
                                                     const float* __restrict__ te) {
  long i = (long)blockIdx.x * 256 + threadIdx.x;
  if (i < (long)NODES * HD) {
    long node = i >> 6;
    hb[i] = (__bf16)te[((node >= NV) ? HD : 0) + (int)(i & 63)];
  }
}

__global__ __launch_bounds__(256) void zero_kernel(float* __restrict__ p, long n) {
  long i = (long)blockIdx.x * 256 + threadIdx.x;
  if (i < n) p[i] = 0.0f;
}

// --------------------------- edge message kernel ---------------------------
// msg = relu([h[iA], h[iB], sign] @ w1.T + b1) @ w2.T + b2 ; atomic add into
// agg[idxDst - dstOff]. One wave handles 16 edges.
__global__ __launch_bounds__(256) void msg_kernel(
    const __bf16* __restrict__ hb,
    const int*    __restrict__ idxA,
    const int*    __restrict__ idxB,
    const int*    __restrict__ idxDst, int dstOff,
    const float*  __restrict__ signs,
    const float*  __restrict__ w1,   // [64][129]
    const float*  __restrict__ b1,   // [64]
    const float*  __restrict__ w2,   // [64][64]
    const float*  __restrict__ b2,   // [64]
    float* __restrict__ agg, int ntiles) {
  __shared__ __bf16 w1s[64 * 128];
  __shared__ __bf16 w2s[64 * 64];
  __shared__ float  w1sg[64];
  __shared__ float  b1s[64];
  __shared__ float  b2s[64];
  __shared__ __bf16 hidS[8][16 * 64];   // layer-1 output staged as bf16

  const int tid = threadIdx.x;
  for (int i = tid; i < 64 * 128; i += 256)
    w1s[i] = (__bf16)w1[(i >> 7) * 129 + (i & 127)];
  for (int i = tid; i < 64 * 64; i += 256) w2s[i] = (__bf16)w2[i];
  if (tid < 64) {
    w1sg[tid] = w1[tid * 129 + 128];
    b1s[tid]  = b1[tid];
    b2s[tid]  = b2[tid];
  }
  __syncthreads();

  const int wave = tid >> 5;
  const int lane = tid & 31;
  const int lo   = lane & 15;
  const int hi   = lane >> 4;
  const int gw   = blockIdx.x * 8 + wave;
  if (gw >= ntiles) return;                 // wave-uniform: EXEC stays all-1s
  const int eb = gw * 16;
  __bf16* hid = hidS[wave];

  const long iA = idxA[eb + lo];
  const long iB = idxB[eb + lo];
  v16bf a1[2], a2[2];
#pragma unroll
  for (int s = 0; s < 2; ++s) {
    a1[s] = a_frag_bf16(hb + iA * HD, s, lane);
    a2[s] = a_frag_bf16(hb + iB * HD, s, lane);
  }
  float sg[8];
#pragma unroll
  for (int r = 0; r < 8; ++r) sg[r] = signs[eb + r + 8 * hi];

  // layer 1: 16 WMMAs (4 N-tiles x 2 K-steps x 2 operand halves)
#pragma unroll
  for (int j = 0; j < 4; ++j) {
    v8f acc = {};
#pragma unroll
    for (int s = 0; s < 2; ++s) {
      acc = wmma_bf16(a1[s], b_frag(w1s, 128, j, s * 32, lane), acc);
      acc = wmma_bf16(a2[s], b_frag(w1s, 128, j, 64 + s * 32, lane), acc);
    }
    const int   n   = j * 16 + lo;
    const float bb  = b1s[n];
    const float wsg = w1sg[n];
#pragma unroll
    for (int r = 0; r < 8; ++r) {
      float v = acc[r] + bb + sg[r] * wsg;      // sign column + bias
      hid[(r + 8 * hi) * HD + n] = (__bf16)fmaxf(v, 0.0f);
    }
  }
  // drain LDS stores before cross-lane reads (wave-synchronous re-layout)
  asm volatile("s_wait_dscnt 0" ::: "memory");

  // layer 2: re-load hidden as A fragments (pure 16B ds loads), 8 WMMAs
  v16bf ah[2];
#pragma unroll
  for (int s = 0; s < 2; ++s) ah[s] = a_frag_bf16(hid + lo * HD, s, lane);

  int dr[8];
#pragma unroll
  for (int r = 0; r < 8; ++r) dr[r] = idxDst[eb + r + 8 * hi] - dstOff;

#pragma unroll
  for (int j = 0; j < 4; ++j) {
    v8f acc = {};
#pragma unroll
    for (int s = 0; s < 2; ++s)
      acc = wmma_bf16(ah[s], b_frag(w2s, 64, j, s * 32, lane), acc);
    const int   n  = j * 16 + lo;
    const float bb = b2s[n];
#pragma unroll
    for (int r = 0; r < 8; ++r)
      atomicAdd(agg + (long)dr[r] * HD + n, acc[r] + bb);
  }
}

// ------------------------------- GRU kernel --------------------------------
// h[hOff+nb+m] = GRU(agg[nb+m], h[hOff+nb+m]); one wave = 16 contiguous nodes.
__global__ __launch_bounds__(256) void gru_kernel(
    __bf16* __restrict__ hb, int hOff,
    const float* __restrict__ agg,
    const float* __restrict__ wih,  // [192][64]
    const float* __restrict__ whh,  // [192][64]
    const float* __restrict__ bih,  // [192]
    const float* __restrict__ bhh,  // [192]
    int ntiles) {
  __shared__ __bf16 wihS[192 * 64];
  __shared__ __bf16 whhS[192 * 64];
  __shared__ float  bihS[192];
  __shared__ float  bhhS[192];

  const int tid = threadIdx.x;
  for (int i = tid; i < 192 * 64; i += 256) {
    wihS[i] = (__bf16)wih[i];
    whhS[i] = (__bf16)whh[i];
  }
  if (tid < 192) { bihS[tid] = bih[tid]; bhhS[tid] = bhh[tid]; }
  __syncthreads();

  const int wave = tid >> 5;
  const int lane = tid & 31;
  const int lo   = lane & 15;
  const int hi   = lane >> 4;
  const int gw   = blockIdx.x * 8 + wave;
  if (gw >= ntiles) return;
  const long nb = (long)gw * 16;

  v16bf ax[2], ah[2];
#pragma unroll
  for (int s = 0; s < 2; ++s) {
    ax[s] = a_frag_f32(agg + (nb + lo) * HD, s, lane);
    ah[s] = a_frag_bf16(hb + (hOff + nb + lo) * HD, s, lane);
  }

  // 48 WMMAs: 3 gates x (gi, gh) x 4 N-tiles x 2 K-steps; fused epilogue per j
#pragma unroll
  for (int j = 0; j < 4; ++j) {
    v8f gr = {}, gz = {}, gn = {}, hr = {}, hz = {}, hn = {};
#pragma unroll
    for (int s = 0; s < 2; ++s) {
      gr = wmma_bf16(ax[s], b_frag(wihS, 64, j,     s * 32, lane), gr);
      gz = wmma_bf16(ax[s], b_frag(wihS, 64, j + 4, s * 32, lane), gz);
      gn = wmma_bf16(ax[s], b_frag(wihS, 64, j + 8, s * 32, lane), gn);
      hr = wmma_bf16(ah[s], b_frag(whhS, 64, j,     s * 32, lane), hr);
      hz = wmma_bf16(ah[s], b_frag(whhS, 64, j + 4, s * 32, lane), hz);
      hn = wmma_bf16(ah[s], b_frag(whhS, 64, j + 8, s * 32, lane), hn);
    }
    const int   col = j * 16 + lo;
    const float bir = bihS[col],       bhr = bhhS[col];
    const float biz = bihS[64 + col],  bhz = bhhS[64 + col];
    const float bin = bihS[128 + col], bhn = bhhS[128 + col];
#pragma unroll
    for (int r = 0; r < 8; ++r) {
      const int m  = r + 8 * hi;
      float rg = sigmoidf_(gr[r] + bir + hr[r] + bhr);
      float zg = sigmoidf_(gz[r] + biz + hz[r] + bhz);
      float ng = tanhf(gn[r] + bin + rg * (hn[r] + bhn));
      __bf16* hp = hb + (hOff + nb + m) * HD + col;
      float hv = (float)*hp;
      *hp = (__bf16)((1.0f - zg) * ng + zg * hv);  // rows owned by this wave
    }
  }
}

// ---------------------------- projection kernel ----------------------------
__global__ __launch_bounds__(256) void proj_kernel(
    const __bf16* __restrict__ hb,
    const float* __restrict__ w1,  // [64][64]
    const float* __restrict__ b1,  // [64]
    const float* __restrict__ w2,  // [64]
    const float* __restrict__ b2,  // [1]
    float* __restrict__ out, int ntiles) {
  __shared__ __bf16 w1S[64 * 64];
  __shared__ float  b1S[64];
  __shared__ float  w2S[64];
  __shared__ float  hidS[8][16 * 64];

  const int tid = threadIdx.x;
  for (int i = tid; i < 64 * 64; i += 256) w1S[i] = (__bf16)w1[i];
  if (tid < 64) { b1S[tid] = b1[tid]; w2S[tid] = w2[tid]; }
  __syncthreads();

  const int wave = tid >> 5;
  const int lane = tid & 31;
  const int lo   = lane & 15;
  const int hi   = lane >> 4;
  const int gw   = blockIdx.x * 8 + wave;
  if (gw >= ntiles) return;
  const long vb = (long)gw * 16;
  float* hid = hidS[wave];

  v16bf a[2];
#pragma unroll
  for (int s = 0; s < 2; ++s) a[s] = a_frag_bf16(hb + (vb + lo) * HD, s, lane);

#pragma unroll
  for (int j = 0; j < 4; ++j) {
    v8f acc = {};
#pragma unroll
    for (int s = 0; s < 2; ++s)
      acc = wmma_bf16(a[s], b_frag(w1S, 64, j, s * 32, lane), acc);
    const int   n  = j * 16 + lo;
    const float bb = b1S[n];
#pragma unroll
    for (int r = 0; r < 8; ++r)
      hid[(r + 8 * hi) * HD + n] = fmaxf(acc[r] + bb, 0.0f);
  }
  asm volatile("s_wait_dscnt 0" ::: "memory");

  if (lane < 16) {
    float acc = b2[0];
#pragma unroll 8
    for (int k = 0; k < 64; ++k) acc += hid[lane * HD + k] * w2S[k];
    out[vb + lane] = sigmoidf_(acc);
  }
}

// ------------------------------ host driver --------------------------------
extern "C" void kernel_launch(void* const* d_in, const int* in_sizes, int n_in,
                              void* d_out, int out_size, void* d_ws, size_t ws_size,
                              hipStream_t stream) {
  (void)in_sizes; (void)n_in; (void)out_size; (void)ws_size;
  const float* te        = (const float*)d_in[0];
  const float* v2c_w1    = (const float*)d_in[1];
  const float* v2c_b1    = (const float*)d_in[2];
  const float* v2c_w2    = (const float*)d_in[3];
  const float* v2c_b2    = (const float*)d_in[4];
  const float* c2v_w1    = (const float*)d_in[5];
  const float* c2v_b1    = (const float*)d_in[6];
  const float* c2v_w2    = (const float*)d_in[7];
  const float* c2v_b2    = (const float*)d_in[8];
  const float* gru_c_wih = (const float*)d_in[9];
  const float* gru_c_whh = (const float*)d_in[10];
  const float* gru_c_bih = (const float*)d_in[11];
  const float* gru_c_bhh = (const float*)d_in[12];
  const float* gru_v_wih = (const float*)d_in[13];
  const float* gru_v_whh = (const float*)d_in[14];
  const float* gru_v_bih = (const float*)d_in[15];
  const float* gru_v_bhh = (const float*)d_in[16];
  const float* proj_w1   = (const float*)d_in[17];
  const float* proj_b1   = (const float*)d_in[18];
  const float* proj_w2   = (const float*)d_in[19];
  const float* proj_b2   = (const float*)d_in[20];
  const float* signs     = (const float*)d_in[21];
  const int*   v_idx     = (const int*)d_in[22];
  const int*   c_idx     = (const int*)d_in[23];

  float*  out = (float*)d_out;
  __bf16* hb  = (__bf16*)d_ws;                               // [NODES][64] bf16
  float*  agg = (float*)((char*)d_ws +
                         (size_t)NODES * HD * sizeof(__bf16)); // [NC][64] f32

  const long nh = (long)NODES * HD;
  init_h_kernel<<<(int)((nh + 255) / 256), 256, 0, stream>>>(hb, te);

  const int etiles  = NEDGE / 16;                  // 56250
  const int eblocks = (etiles + 7) / 8;
  const int ctiles  = NC / 16;                     // 18750
  const int cblocks = (ctiles + 7) / 8;
  const int vtiles  = NV / 16;                     // 6250
  const int vblocks = (vtiles + 7) / 8;

  for (int layer = 0; layer < 4; ++layer) {
    // variable -> clause
    const long nac = (long)NC * HD;
    zero_kernel<<<(int)((nac + 255) / 256), 256, 0, stream>>>(agg, nac);
    msg_kernel<<<eblocks, 256, 0, stream>>>(hb, v_idx, c_idx, c_idx, NV, signs,
                                            v2c_w1, v2c_b1, v2c_w2, v2c_b2,
                                            agg, etiles);
    gru_kernel<<<cblocks, 256, 0, stream>>>(hb, NV, agg,
                                            gru_c_wih, gru_c_whh,
                                            gru_c_bih, gru_c_bhh, ctiles);
    // clause -> variable
    const long nav = (long)NV * HD;
    zero_kernel<<<(int)((nav + 255) / 256), 256, 0, stream>>>(agg, nav);
    msg_kernel<<<eblocks, 256, 0, stream>>>(hb, c_idx, v_idx, v_idx, 0, signs,
                                            c2v_w1, c2v_b1, c2v_w2, c2v_b2,
                                            agg, etiles);
    gru_kernel<<<vblocks, 256, 0, stream>>>(hb, 0, agg,
                                            gru_v_wih, gru_v_whh,
                                            gru_v_bih, gru_v_bhh, vtiles);
  }

  proj_kernel<<<vblocks, 256, 0, stream>>>(hb, proj_w1, proj_b1, proj_w2, proj_b2,
                                           out, vtiles);
}
